// RelPosAttention_32667521253707
// MI455X (gfx1250) — matrix operand order repaired and checked by reading, not compiled, not run on hardware
//
#include <hip/hip_runtime.h>
#include <hip/hip_bf16.h>

#define DIM 768
#define HEADS 12
#define HEAD_DIM 64
#define BATCH 8
#define SEQ 1024

typedef __attribute__((ext_vector_type(8)))  __bf16 v8bf;
typedef __attribute__((ext_vector_type(16))) __bf16 v16bf;
typedef __attribute__((ext_vector_type(8)))  float  v8f;
typedef __attribute__((ext_vector_type(4)))  int    v4i;
typedef __attribute__((address_space(1))) v4i AS1v4i;
typedef __attribute__((address_space(3))) v4i AS3v4i;

union Frag { v16bf v; v8bf h[2]; };

// A-matrix 16x32 bf16 fragment (row selected by caller via rp):
// lane<16 holds K {0..7,16..23}, lane>=16 holds K {8..15,24..31} (ISA 7.12.2)
__device__ __forceinline__ v16bf load_a_frag(const __bf16* rp, int lane, int kbase) {
  int c0 = kbase + ((lane >> 4) << 3);
  Frag f;
  f.h[0] = *reinterpret_cast<const v8bf*>(rp + c0);
  f.h[1] = *reinterpret_cast<const v8bf*>(rp + c0 + 16);
  return f.v;
}

// B-matrix 32x16 bf16 fragment (column selected by caller via cp):
// lane<16 holds K {0..7,16..23}, lane>=16 holds K {8..15,24..31}
__device__ __forceinline__ v16bf load_b_frag(const __bf16* cp, int lane, int kbase) {
  int c0 = kbase + ((lane >> 4) << 3);
  Frag f;
  f.h[0] = *reinterpret_cast<const v8bf*>(cp + c0);
  f.h[1] = *reinterpret_cast<const v8bf*>(cp + c0 + 16);
  return f.v;
}

// 16B global -> LDS copy on the async-DMA path (ASYNCcnt) when available.
__device__ __forceinline__ void cp_async16(const __bf16* g, __bf16* l) {
#if __has_builtin(__builtin_amdgcn_global_load_async_to_lds_b128)
  __builtin_amdgcn_global_load_async_to_lds_b128((AS1v4i*)g, (AS3v4i*)l, 0, 0);
#else
  *reinterpret_cast<v8bf*>(l) = *reinterpret_cast<const v8bf*>(g);
#endif
}

__device__ __forceinline__ void async_wait0() {
  asm volatile("s_wait_asynccnt 0x0" ::: "memory");
}

__global__ void cvt_f32_bf16(const float* __restrict__ src, __bf16* __restrict__ dst, int n) {
  int i = blockIdx.x * blockDim.x + threadIdx.x;
  if (i < n) dst[i] = (__bf16)src[i];
}

// Double-buffered staged-LDS GEMM: workgroup tile 64 rows x 128 cols, K step 32.
// Async DMA for step i+1 overlaps the WMMAs of step i; one barrier per step.
// MODE 0: QKV epilogue (scatter q*scale / k / v^T bf16). MODE 1: proj (+bias, f32 out).
template <int MODE>
__global__ void __launch_bounds__(256)
gemm_stage(const __bf16* __restrict__ X, const __bf16* __restrict__ W,
           __bf16* __restrict__ Q, __bf16* __restrict__ Kq, __bf16* __restrict__ VT,
           const float* __restrict__ pbias, float* __restrict__ pout) {
  __shared__ __bf16 As[2][64 * 32];
  __shared__ __bf16 Bs[2][128 * 32];
  const int tid = threadIdx.x;
  const int lane = tid & 31, wv = tid >> 5;
  const int m0l = (wv >> 2) * 32;  // wave tile origin inside workgroup tile
  const int o0l = (wv & 3) * 32;
  const int mblk = blockIdx.x * 64;
  const int oblk = blockIdx.y * 128;

  auto stage = [&](int buf, int k0) {
    int row = tid >> 2, off = (tid & 3) * 8;  // A: 64x32, one 16B chunk/thread
    cp_async16(X + (size_t)(mblk + row) * DIM + k0 + off, As[buf] + row * 32 + off);
#pragma unroll
    for (int c = 0; c < 2; ++c) {             // B: 128x32, two chunks/thread
      int idx = tid + c * 256;
      int r2 = idx >> 2, o2 = (idx & 3) * 8;
      cp_async16(W + (size_t)(oblk + r2) * DIM + k0 + o2, Bs[buf] + r2 * 32 + o2);
    }
  };

  stage(0, 0);
  v8f acc[2][2] = {};
  for (int k0 = 0; k0 < DIM; k0 += 32) {
    const int cur = (k0 >> 5) & 1;
    async_wait0();      // only the current buffer's DMA is outstanding
    __syncthreads();    // all waves finished reading the other buffer last step
    if (k0 + 32 < DIM) stage(cur ^ 1, k0 + 32);  // overlaps with WMMAs below

    v16bf a[2], b[2];
#pragma unroll
    for (int i = 0; i < 2; ++i) {
      a[i] = load_a_frag(As[cur] + (size_t)(m0l + i * 16 + (lane & 15)) * 32, lane, 0);
      b[i] = load_b_frag(Bs[cur] + (size_t)(o0l + i * 16 + (lane & 15)) * 32, lane, 0);
    }
#pragma unroll
    for (int i = 0; i < 2; ++i)
#pragma unroll
      for (int j = 0; j < 2; ++j)
        acc[i][j] = __builtin_amdgcn_wmma_f32_16x16x32_bf16(
            false, a[i], false, b[j], (short)0, acc[i][j], false, false);
  }

  const int m0 = mblk + m0l, o0 = oblk + o0l;
#pragma unroll
  for (int i = 0; i < 2; ++i) {
    int row0 = m0 + i * 16 + ((lane >> 4) << 3);
#pragma unroll
    for (int j = 0; j < 2; ++j) {
      int o = o0 + j * 16 + (lane & 15);
      if (MODE == 0) {
        int three = o / DIM;  // uniform per 16-col tile (768 % 16 == 0)
        int rem = o % DIM;
        int hh = rem / HEAD_DIM, d = rem % HEAD_DIM;
#pragma unroll
        for (int r = 0; r < 8; ++r) {
          int mrow = row0 + r;
          int bidx = mrow >> 10, n = mrow & (SEQ - 1);
          size_t qk = ((size_t)(bidx * HEADS + hh) * SEQ + n) * HEAD_DIM + d;
          float vv = acc[i][j][r];
          if (three == 0)      Q[qk] = (__bf16)(vv * 0.125f);  // HEAD_DIM^-0.5
          else if (three == 1) Kq[qk] = (__bf16)vv;
          else VT[((size_t)(bidx * HEADS + hh) * HEAD_DIM + d) * SEQ + n] = (__bf16)vv;
        }
      } else {
        float bv = pbias[o];
#pragma unroll
        for (int r = 0; r < 8; ++r)
          pout[(size_t)(row0 + r) * DIM + o] = acc[i][j][r] + bv;
      }
    }
  }
}

// Flash attention: 4 waves / workgroup, each wave owns 16 query rows of one (b,h).
// K/V key-blocks double-buffered through async-to-LDS; S initialized from the
// rel-pos bias (WMMA C operand); P transposed D-layout->A-layout through LDS.
__global__ void __launch_bounds__(128)
attn_flash(const __bf16* __restrict__ Q, const __bf16* __restrict__ K,
           const __bf16* __restrict__ VT, const float* __restrict__ bias,
           __bf16* __restrict__ O) {
  __shared__ __bf16 Kt[2][32 * 64];   // keys x head_dim (row-major, as in global)
  __shared__ __bf16 Vt[2][64 * 32];   // head_dim x keys (rows of V^T)
  __shared__ __bf16 Pl[4][16 * 32];   // per-wave P transpose scratch
  const int lane = threadIdx.x & 31;
  const int wv   = threadIdx.x >> 5;
  const int tid  = threadIdx.x;
  const int bh = blockIdx.y;
  const int b = bh / HEADS, h = bh % HEADS;
  const int q0 = blockIdx.x * 64 + wv * 16;

  const __bf16* qbase  = Q  + (size_t)bh * SEQ * HEAD_DIM;
  const __bf16* kbase  = K  + (size_t)bh * SEQ * HEAD_DIM;
  const __bf16* vtbase = VT + (size_t)bh * HEAD_DIM * SEQ;
  const float*  bbase  = bias + (size_t)h * SEQ * SEQ;
  __bf16* plds = Pl[wv];

  auto stageKV = [&](int buf, int kt2) {
#pragma unroll
    for (int c = 0; c < 2; ++c) {
      int idx = tid + c * 128;               // K block: contiguous 2048 bf16
      cp_async16(kbase + (size_t)kt2 * HEAD_DIM + idx * 8, Kt[buf] + idx * 8);
      int d = idx >> 2, off = (idx & 3) * 8; // V^T rows: 64B segments per channel
      cp_async16(vtbase + (size_t)d * SEQ + kt2 + off, Vt[buf] + d * 32 + off);
    }
  };

  v16bf qa[2];
  {
    const __bf16* rp = qbase + (size_t)(q0 + (lane & 15)) * HEAD_DIM;
    qa[0] = load_a_frag(rp, lane, 0);
    qa[1] = load_a_frag(rp, lane, 32);
  }

  v8f oacc[4] = {};
  float mi[8], li[8];
#pragma unroll
  for (int r = 0; r < 8; ++r) { mi[r] = -__builtin_inff(); li[r] = 0.f; }

  const int rbase = q0 + ((lane >> 4) << 3);  // first of 8 C/D rows this lane owns

  stageKV(0, 0);
  for (int kt = 0; kt < SEQ; kt += 32) {
    const int cur = (kt >> 5) & 1;

    // ---- bias streams from global (L2-resident), hiding the DMA tail ----
    v8f s[2];
#pragma unroll
    for (int t = 0; t < 2; ++t)
#pragma unroll
      for (int r = 0; r < 8; ++r)
        s[t][r] = bbase[(size_t)(rbase + r) * SEQ + kt + t * 16 + (lane & 15)];
    if (kt + 32 < SEQ) {
#pragma unroll
      for (int r = 0; r < 8; ++r)
        __builtin_prefetch(bbase + (size_t)(rbase + r) * SEQ + kt + 32 + (lane & 15), 0, 1);
    }

    async_wait0();      // current K/V block landed in LDS
    __syncthreads();    // everyone done reading the other buffer last step
    if (kt + 32 < SEQ) stageKV(cur ^ 1, kt + 32);  // next block DMA under compute

    // ---- S = bias + q k^T ----
#pragma unroll
    for (int t = 0; t < 2; ++t) {
      const __bf16* cp = Kt[cur] + (size_t)(t * 16 + (lane & 15)) * HEAD_DIM;
#pragma unroll
      for (int kk = 0; kk < 2; ++kk) {
        v16bf kf = load_b_frag(cp, lane, kk * 32);
        s[t] = __builtin_amdgcn_wmma_f32_16x16x32_bf16(
            false, qa[kk], false, kf, (short)0, s[t], false, false);
      }
    }

    // ---- online softmax (xor shuffles stay inside each 16-lane row-half) ----
    float mnew[8], alpha[8];
#pragma unroll
    for (int r = 0; r < 8; ++r) {
      float v = fmaxf(s[0][r], s[1][r]);
      v = fmaxf(v, __shfl_xor(v, 1, 32));
      v = fmaxf(v, __shfl_xor(v, 2, 32));
      v = fmaxf(v, __shfl_xor(v, 4, 32));
      v = fmaxf(v, __shfl_xor(v, 8, 32));
      mnew[r]  = fmaxf(mi[r], v);
      alpha[r] = __expf(mi[r] - mnew[r]);
    }
#pragma unroll
    for (int t = 0; t < 2; ++t)
#pragma unroll
      for (int r = 0; r < 8; ++r)
        s[t][r] = __expf(s[t][r] - mnew[r]);
#pragma unroll
    for (int r = 0; r < 8; ++r) {
      float v = s[0][r] + s[1][r];
      v += __shfl_xor(v, 1, 32);
      v += __shfl_xor(v, 2, 32);
      v += __shfl_xor(v, 4, 32);
      v += __shfl_xor(v, 8, 32);
      li[r] = li[r] * alpha[r] + v;
      mi[r] = mnew[r];
    }
#pragma unroll
    for (int td = 0; td < 4; ++td)
#pragma unroll
      for (int r = 0; r < 8; ++r)
        oacc[td][r] *= alpha[r];

    // ---- P: C/D layout -> A-fragment layout via per-wave LDS (same-wave in-order) ----
#pragma unroll
    for (int t = 0; t < 2; ++t)
#pragma unroll
      for (int r = 0; r < 8; ++r)
        plds[(r + ((lane >> 4) << 3)) * 32 + t * 16 + (lane & 15)] = (__bf16)s[t][r];

    v16bf pa;
    {
      const __bf16* rp = plds + (size_t)(lane & 15) * 32;
      int c0 = (lane >> 4) << 3;
      Frag f;
      f.h[0] = *reinterpret_cast<const v8bf*>(rp + c0);
      f.h[1] = *reinterpret_cast<const v8bf*>(rp + c0 + 16);
      pa = f.v;
    }

    // ---- O += P x V (V^T rows in LDS -> contiguous-K B fragments) ----
#pragma unroll
    for (int td = 0; td < 4; ++td) {
      const __bf16* cp = Vt[cur] + (size_t)(td * 16 + (lane & 15)) * 32;
      v16bf vf = load_b_frag(cp, lane, 0);
      oacc[td] = __builtin_amdgcn_wmma_f32_16x16x32_bf16(
          false, pa, false, vf, (short)0, oacc[td], false, false);
    }
  }

#pragma unroll
  for (int r = 0; r < 8; ++r) {
    float inv = 1.0f / li[r];
    size_t tok = (size_t)b * SEQ + rbase + r;
#pragma unroll
    for (int td = 0; td < 4; ++td)
      O[tok * DIM + h * HEAD_DIM + td * 16 + (lane & 15)] = (__bf16)(oacc[td][r] * inv);
  }
}

extern "C" void kernel_launch(void* const* d_in, const int* in_sizes, int n_in,
                              void* d_out, int out_size, void* d_ws, size_t ws_size,
                              hipStream_t stream) {
  const float* x      = (const float*)d_in[0];
  const float* relpos = (const float*)d_in[1];
  const float* qkv_w  = (const float*)d_in[2];
  const float* proj_w = (const float*)d_in[3];
  const float* proj_b = (const float*)d_in[4];
  float* out = (float*)d_out;

  char* ws = (char*)d_ws;
  size_t off = 0;
  auto alloc = [&](size_t bytes) {
    void* p = ws + off;
    off += (bytes + 255) & ~(size_t)255;
    return p;
  };
  const int ntok = BATCH * SEQ;
  __bf16* xb  = (__bf16*)alloc((size_t)ntok * DIM * 2);
  __bf16* wqb = (__bf16*)alloc((size_t)3 * DIM * DIM * 2);
  __bf16* wpb = (__bf16*)alloc((size_t)DIM * DIM * 2);
  __bf16* Qb  = (__bf16*)alloc((size_t)ntok * DIM * 2);
  __bf16* Kb  = (__bf16*)alloc((size_t)ntok * DIM * 2);
  __bf16* VTb = (__bf16*)alloc((size_t)ntok * DIM * 2);
  __bf16* AOb = (__bf16*)alloc((size_t)ntok * DIM * 2);

  int n1 = ntok * DIM;
  cvt_f32_bf16<<<(n1 + 255) / 256, 256, 0, stream>>>(x, xb, n1);
  int n2 = 3 * DIM * DIM;
  cvt_f32_bf16<<<(n2 + 255) / 256, 256, 0, stream>>>(qkv_w, wqb, n2);
  int n3 = DIM * DIM;
  cvt_f32_bf16<<<(n3 + 255) / 256, 256, 0, stream>>>(proj_w, wpb, n3);

  gemm_stage<0><<<dim3(ntok / 64, (3 * DIM) / 128), 256, 0, stream>>>(
      xb, wqb, Qb, Kb, VTb, nullptr, nullptr);
  attn_flash<<<dim3(SEQ / 64, BATCH * HEADS), 128, 0, stream>>>(Qb, Kb, VTb, relpos, AOb);
  gemm_stage<1><<<dim3(ntok / 64, DIM / 128), 256, 0, stream>>>(
      AOb, wpb, nullptr, nullptr, nullptr, proj_b, out);
}
